// LocalTemporalSlidingAttention_88424786690218
// MI455X (gfx1250) — matrix operand order repaired and checked by reading, not compiled, not run on hardware
//
#include <hip/hip_runtime.h>
#include <hip/hip_bf16.h>

#define BB   32
#define FF   64
#define MM   8
#define TT   1024
#define KK   32            // conv taps
#define EE   (FF*MM)       // 512
#define WIN  32
#define P2   4
#define EPS  1e-5f
#define ASCALE 0.125f      // 1/sqrt(F)
#define WPB  4             // waves per block in attention kernel
#define NKT  5             // live key tiles per 16-query tile (keys s-32 .. s+47)

typedef __attribute__((ext_vector_type(16))) _Float16 v16h;
typedef __attribute__((ext_vector_type(8)))  float    v8f;
typedef __attribute__((ext_vector_type(4)))  float    v4f;

static __device__ __forceinline__ v8f zero8() {
  v8f z;
#pragma unroll
  for (int i = 0; i < 8; ++i) z[i] = 0.0f;
  return z;
}

// ---------------------------------------------------------------- init stats
__global__ void init_stats_kernel(float* stats) {
  if (threadIdx.x < 256) stats[threadIdx.x] = 0.0f;
}

// --------------------------- depthwise QKV convs + fused q/k transpose
// block = (b, 32-e x 128-t tile). Outputs:
//   q,k : [B,T,E] f16 (transposed through LDS, 32B-contiguous stores)
//   v   : [B,E,T] f16 (natural, 32B-contiguous stores)
#define QE 32
#define QT 128
__global__ __launch_bounds__(256) void qkv_kernel(
    const float* __restrict__ x,
    const float* __restrict__ wq, const float* __restrict__ bq,
    const float* __restrict__ wk, const float* __restrict__ bk,
    const float* __restrict__ wv, const float* __restrict__ bv,
    _Float16* __restrict__ qt, _Float16* __restrict__ ktr, _Float16* __restrict__ vn) {
  __shared__ float xs[QE][QT + KK + 1];          // [32][161] halo'd input
  __shared__ float wql[4][KK], wkl[4][KK], wvl[4][KK];
  __shared__ _Float16 qs[QE][QT], ks[QE][QT];    // staged for transpose
  const int tid = threadIdx.x;
  const int nt = TT / QT;                        // 8
  const int ne = EE / QE;                        // 16
  const int b  = blockIdx.x / (ne * nt);
  const int r  = blockIdx.x % (ne * nt);
  const int e0 = (r / nt) * QE;
  const int t0 = (r % nt) * QT;
  // load x halo [t0-15, t0+QT+15]
  for (int idx = tid; idx < QE * (QT + KK); idx += 256) {
    int el = idx / (QT + KK);
    int ii = idx % (QT + KK);
    int t  = t0 - 15 + ii;
    xs[el][ii] = (ii < QT + KK - 1 && t >= 0 && t < TT)
                 ? x[((size_t)b * EE + e0 + el) * TT + t] : 0.0f;
  }
  if (tid < 4 * KK) {                            // 4 f-channels per block
    int fl = tid >> 5, kk = tid & 31;
    int f = (e0 >> 3) + fl;
    wql[fl][kk] = wq[f * KK + kk];
    wkl[fl][kk] = wk[f * KK + kk];
    wvl[fl][kk] = wv[f * KK + kk];
  }
  __syncthreads();
  // compute: thread handles (e-row el, 16 consecutive t)
  {
    const int el = tid >> 3;
    const int ts = (tid & 7) * 16;
    const int fl = el >> 3;
    const int f  = (e0 >> 3) + fl;
    const float cq = bq[f], ck = bk[f], cv = bv[f];
    _Float16 vloc[16];
#pragma unroll
    for (int u = 0; u < 16; ++u) {
      float sq = cq, sk = ck, sv = cv;
#pragma unroll
      for (int kk = 0; kk < KK; ++kk) {
        float xv = xs[el][ts + u + kk];
        sq = fmaf(xv, wql[fl][kk], sq);
        sk = fmaf(xv, wkl[fl][kk], sk);
        sv = fmaf(xv, wvl[fl][kk], sv);
      }
      qs[el][ts + u] = (_Float16)sq;
      ks[el][ts + u] = (_Float16)sk;
      vloc[u] = (_Float16)sv;
    }
    _Float16* vp = vn + ((size_t)b * EE + e0 + el) * TT + t0 + ts;
#pragma unroll
    for (int u = 0; u < 16; ++u) vp[u] = vloc[u];
  }
  __syncthreads();
  // transposed stores: thread handles (t-row tl, 16 consecutive e)
  {
    const int tl = tid >> 1;
    const int ec = (tid & 1) * 16;
    _Float16* qp = qt  + ((size_t)b * TT + t0 + tl) * EE + e0 + ec;
    _Float16* kp = ktr + ((size_t)b * TT + t0 + tl) * EE + e0 + ec;
#pragma unroll
    for (int j = 0; j < 16; ++j) {
      qp[j] = qs[ec + j][tl];
      kp[j] = ks[ec + j][tl];
    }
  }
}

// ------------------------------------------------------- banded WMMA attention
// one wave per (b, 16-query tile); attn out: [B,E,T] f32
__global__ __launch_bounds__(32 * WPB) void attn_kernel(
    const _Float16* __restrict__ q, const _Float16* __restrict__ k,
    const _Float16* __restrict__ v, float* __restrict__ attn) {
  __shared__ _Float16 P[WPB][16][96];     // softmaxed band probabilities (cols 80..95 zero)
  __shared__ float stag[WPB][16][17];     // 16x16 output staging (padded)
  const int lane = threadIdx.x & 31;
  const int wv   = threadIdx.x >> 5;
  const int tile = blockIdx.x * WPB + wv;
  const int b    = tile / (TT / 16);
  const int s    = (tile % (TT / 16)) * 16;
  const int half = lane >> 4;
  const int n    = lane & 15;             // B/C/D column, also A row

  // ---- scores: 5 key tiles of 16 (keys s-32 .. s+47), K-dim = E = 512
  v8f acc[NKT];
#pragma unroll
  for (int kt = 0; kt < NKT; ++kt) acc[kt] = zero8();

  const _Float16* qb = q + ((size_t)b * TT + (s + n)) * EE;   // A row m = lane&15
  const _Float16* kbp[NKT];
#pragma unroll
  for (int kt = 0; kt < NKT; ++kt) {
    int key  = s - 32 + kt * 16 + n;
    int keyc = min(max(key, 0), TT - 1);                      // masked later anyway
    kbp[kt] = k + ((size_t)b * TT + keyc) * EE + half * 16;
  }
  for (int kc = 0; kc < 16; ++kc) {
    v16h a;
    const _Float16* ap = qb + kc * 32 + half * 16;            // A K-pattern: K = h*16 + j
#pragma unroll
    for (int j = 0; j < 16; ++j) a[j] = ap[j];
    v16h bf[NKT];                                             // batch all B loads first
#pragma unroll
    for (int kt = 0; kt < NKT; ++kt) {
      const _Float16* bp = kbp[kt] + kc * 32;                 // B K-pattern: K = h*16 + j
#pragma unroll
      for (int j = 0; j < 16; ++j) bf[kt][j] = bp[j];
    }
#pragma unroll
    for (int kt = 0; kt < NKT; ++kt)
      acc[kt] = __builtin_amdgcn_wmma_f32_16x16x32_f16(
          false, a, false, bf[kt], (short)0, acc[kt], false, false);
  }

  // ---- mask (sliding window + bounds): keep iff -32 <= j-i <= 31
#pragma unroll
  for (int r = 0; r < 8; ++r) {
    int irow = s + r + 8 * half;                              // C/D: M = r + 8*(lane/16)
#pragma unroll
    for (int kt = 0; kt < NKT; ++kt) {
      int j = s - 32 + kt * 16 + n;
      int d = j - irow;
      if (!((d >= -WIN) && (d < WIN) && (j >= 0) && (j < TT)))
        acc[kt][r] = -3.0e38f;
    }
  }

  // ---- softmax(scale * scores) per row; rows span 16 lanes x 5 tiles
  float pscale[8];
#pragma unroll
  for (int r = 0; r < 8; ++r) {
    float mx = -3.0e38f;
#pragma unroll
    for (int kt = 0; kt < NKT; ++kt) mx = fmaxf(mx, acc[kt][r]);
    for (int off = 1; off < 16; off <<= 1) mx = fmaxf(mx, __shfl_xor(mx, off, 32));
    float sum = 0.0f;
#pragma unroll
    for (int kt = 0; kt < NKT; ++kt) {
      float p = __expf((acc[kt][r] - mx) * ASCALE);           // masked -> underflow -> 0
      acc[kt][r] = p;
      sum += p;
    }
    for (int off = 1; off < 16; off <<= 1) sum += __shfl_xor(sum, off, 32);
    pscale[r] = 1.0f / sum;
  }

  // ---- stage P (f16) into LDS row-major [m][key]; zero the dead 80..95 cols
#pragma unroll
  for (int r = 0; r < 8; ++r) {
    int mrow = r + 8 * half;
#pragma unroll
    for (int kt = 0; kt < NKT; ++kt)
      P[wv][mrow][kt * 16 + n] = (_Float16)(acc[kt][r] * pscale[r]);
    P[wv][mrow][80 + n] = (_Float16)0.0f;
  }
  __syncthreads();

  // ---- reload P as A fragments (3 chunks of K=32)
  v16h afr[3];
#pragma unroll
  for (int c = 0; c < 3; ++c) {
    const _Float16* pp = &P[wv][n][c * 32 + half * 16];
#pragma unroll
    for (int j = 0; j < 16; ++j) afr[c][j] = pp[j];
  }

  // ---- A·V : 32 output column tiles of 16 features
  for (int et = 0; et < 32; ++et) {
    const int e0 = et * 16;
    v8f acc2 = zero8();
#pragma unroll
    for (int c = 0; c < 3; ++c) {
      int kk0 = s - 32 + c * 32 + half * 16;                  // 16-aligned run, clamp whole run
      kk0 = min(max(kk0, 0), TT - 16);                        // P==0 where clamped/masked
      const _Float16* vp = v + ((size_t)b * EE + (e0 + n)) * TT + kk0;
      v16h bf;
#pragma unroll
      for (int j = 0; j < 16; ++j) bf[j] = vp[j];
      acc2 = __builtin_amdgcn_wmma_f32_16x16x32_f16(
          false, afr[c], false, bf, (short)0, acc2, false, false);
    }
    // stage [t][e] tile, then coalesced store to attn[B,E,T]
#pragma unroll
    for (int r = 0; r < 8; ++r) stag[wv][r + 8 * half][n] = acc2[r];
    __syncthreads();
    const int el = lane >> 1, ts = lane & 1;
    float* op = attn + ((size_t)b * EE + e0 + el) * TT + s + ts * 8;
#pragma unroll
    for (int j = 0; j < 8; ++j) op[j] = stag[wv][ts * 8 + j][el];
    __syncthreads();
  }
}

// ------------------------------------- y1 = hidden + conv_o1(attn), stats1
__global__ __launch_bounds__(256) void conv1_kernel(
    const float* __restrict__ hidden, const float* __restrict__ attn,
    const float* __restrict__ wo1, const float* __restrict__ bo1,
    float* __restrict__ y1, float* __restrict__ stats) {
  __shared__ float row[TT + KK];
  __shared__ float wl[KK];
  __shared__ float red[256];
  const int tid = threadIdx.x;
  const int b = blockIdx.x / EE;
  const int e = blockIdx.x % EE;
  const int f = e >> 3;
  const float* ar = attn + ((size_t)b * EE + e) * TT;
  const float* hr = hidden + ((size_t)b * EE + e) * TT;
  for (int idx = tid; idx < TT + KK - 1; idx += 256) {
    int t = idx - 15;
    row[idx] = (t >= 0 && t < TT) ? ar[t] : 0.0f;
  }
  if (tid < KK) wl[tid] = wo1[f * KK + tid];
  __syncthreads();
  const float bias = bo1[f];
  float ls = 0.0f, lss = 0.0f;
  const int t0 = tid * 4;
  v4f yv;
#pragma unroll
  for (int u = 0; u < 4; ++u) {
    float sy = bias;
#pragma unroll
    for (int kk = 0; kk < KK; ++kk) sy = fmaf(row[t0 + u + kk], wl[kk], sy);
    sy += hr[t0 + u];
    yv[u] = sy;
    ls += sy;
    lss = fmaf(sy, sy, lss);
  }
  *(v4f*)(y1 + ((size_t)b * EE + e) * TT + t0) = yv;
  red[tid] = ls; __syncthreads();
  for (int o = 128; o > 0; o >>= 1) { if (tid < o) red[tid] += red[tid + o]; __syncthreads(); }
  if (tid == 0) atomicAdd(&stats[f], red[0]);
  __syncthreads();
  red[tid] = lss; __syncthreads();
  for (int o = 128; o > 0; o >>= 1) { if (tid < o) red[tid] += red[tid + o]; __syncthreads(); }
  if (tid == 0) atomicAdd(&stats[64 + f], red[0]);
}

// ---------------------------- y2 = conv_o2(bn1(y1)), stats2 (writes into y2)
__global__ __launch_bounds__(256) void conv2_kernel(
    const float* __restrict__ y1,
    const float* __restrict__ wo2, const float* __restrict__ bo2,
    const float* __restrict__ g1, const float* __restrict__ be1,
    float* __restrict__ stats, float* __restrict__ y2) {
  __shared__ float row[TT + KK];
  __shared__ float wl[KK];
  __shared__ float red[256];
  const int tid = threadIdx.x;
  const int b = blockIdx.x / EE;
  const int e = blockIdx.x % EE;
  const int f = e >> 3;
  const float Ninv = 1.0f / (float)(BB * MM * TT);
  const float mu = stats[f] * Ninv;
  const float var = stats[64 + f] * Ninv - mu * mu;
  const float a1 = rsqrtf(var + EPS) * g1[f];
  const float c1 = be1[f] - mu * a1;
  const float* yr = y1 + ((size_t)b * EE + e) * TT;
  for (int idx = tid; idx < TT + KK - 1; idx += 256) {
    int t = idx - 15;
    row[idx] = (t >= 0 && t < TT) ? fmaf(yr[t], a1, c1) : 0.0f;   // bn1 folded into load
  }
  if (tid < KK) wl[tid] = wo2[f * KK + tid];
  __syncthreads();
  const float bias = bo2[f];
  float ls = 0.0f, lss = 0.0f;
  const int t0 = tid * 4;
  v4f yv;
#pragma unroll
  for (int u = 0; u < 4; ++u) {
    float sy = bias;
#pragma unroll
    for (int kk = 0; kk < KK; ++kk) sy = fmaf(row[t0 + u + kk], wl[kk], sy);
    yv[u] = sy;
    ls += sy;
    lss = fmaf(sy, sy, lss);
  }
  *(v4f*)(y2 + ((size_t)b * EE + e) * TT + t0) = yv;
  red[tid] = ls; __syncthreads();
  for (int o = 128; o > 0; o >>= 1) { if (tid < o) red[tid] += red[tid + o]; __syncthreads(); }
  if (tid == 0) atomicAdd(&stats[128 + f], red[0]);
  __syncthreads();
  red[tid] = lss; __syncthreads();
  for (int o = 128; o > 0; o >>= 1) { if (tid < o) red[tid] += red[tid + o]; __syncthreads(); }
  if (tid == 0) atomicAdd(&stats[192 + f], red[0]);
}

// ------------------------------------------- out = avgpool4(bn2(y2))
__global__ __launch_bounds__(256) void finalize_kernel(
    const float* __restrict__ y2, const float* __restrict__ g2,
    const float* __restrict__ be2, const float* __restrict__ stats,
    float* __restrict__ out) {
  const int idx = blockIdx.x * 256 + threadIdx.x;          // [B, E, T/4] flat
  const int TP = TT / P2;
  const int tp = idx % TP;
  const int e  = (idx / TP) % EE;
  const int b  = idx / (TP * EE);
  const int f  = e >> 3;
  const float Ninv = 1.0f / (float)(BB * MM * TT);
  const float mu = stats[128 + f] * Ninv;
  const float var = stats[192 + f] * Ninv - mu * mu;
  const float a2 = rsqrtf(var + EPS) * g2[f];
  const float c2 = be2[f] - mu * a2;
  const float* yr = y2 + ((size_t)b * EE + e) * TT + tp * P2;
  float s = 0.0f;
#pragma unroll
  for (int p = 0; p < P2; ++p) s += fmaf(yr[p], a2, c2);
  out[idx] = s * 0.25f;
}

// -----------------------------------------------------------------------------
extern "C" void kernel_launch(void* const* d_in, const int* in_sizes, int n_in,
                              void* d_out, int out_size, void* d_ws, size_t ws_size,
                              hipStream_t stream) {
  const float* hidden = (const float*)d_in[0];
  const float* wq  = (const float*)d_in[1];  const float* bq  = (const float*)d_in[2];
  const float* wk  = (const float*)d_in[3];  const float* bk  = (const float*)d_in[4];
  const float* wvw = (const float*)d_in[5];  const float* bv  = (const float*)d_in[6];
  const float* wo1 = (const float*)d_in[7];  const float* bo1 = (const float*)d_in[8];
  const float* wo2 = (const float*)d_in[9];  const float* bo2 = (const float*)d_in[10];
  const float* g1  = (const float*)d_in[11]; const float* be1 = (const float*)d_in[12];
  const float* g2  = (const float*)d_in[13]; const float* be2 = (const float*)d_in[14];
  float* out = (float*)d_out;

  char* ws = (char*)d_ws;
  const size_t qB = (size_t)BB * TT * EE * sizeof(_Float16);   // 32 MB
  const size_t fB = (size_t)BB * EE * TT * sizeof(float);      // 64 MB
  _Float16* qt   = (_Float16*)(ws);                            // [B,T,E] f16
  _Float16* ktr  = (_Float16*)(ws + qB);                       // [B,T,E] f16
  _Float16* vbuf = (_Float16*)(ws + 2 * qB);                   // [B,E,T] f16
  float*    attn = (float*)(ws + 3 * qB);                      // attn out; reused as y2
  float*    y1   = (float*)(ws);                               // overlays qt/ktr (dead by then)
  float*    stats= (float*)(ws + 3 * qB + fB);                 // 256 floats
  // total workspace: 3*32MB + 64MB + 1KB = 160MB

  init_stats_kernel<<<1, 256, 0, stream>>>(stats);
  qkv_kernel<<<BB * (EE / QE) * (TT / QT), 256, 0, stream>>>(hidden, wq, bq, wk, bk,
                                                             wvw, bv, qt, ktr, vbuf);
  attn_kernel<<<(BB * (TT / 16)) / WPB, 32 * WPB, 0, stream>>>(qt, ktr, vbuf, attn);
  conv1_kernel<<<BB * EE, 256, 0, stream>>>(hidden, attn, wo1, bo1, y1, stats);
  conv2_kernel<<<BB * EE, 256, 0, stream>>>(y1, wo2, bo2, g1, be1, stats, attn);
  finalize_kernel<<<(BB * EE * (TT / P2)) / 256, 256, 0, stream>>>(attn, g2, be2,
                                                                   stats, out);
}
